// Attention_45913200394506
// MI455X (gfx1250) — matrix operand order repaired and checked by reading, not compiled, not run on hardware
//
#include <hip/hip_runtime.h>

typedef _Float16 f16;
typedef __attribute__((ext_vector_type(16))) _Float16 v16h;
typedef __attribute__((ext_vector_type(8)))  _Float16 v8h;
typedef __attribute__((ext_vector_type(8)))  float    v8f;

#define DIM    1024
#define NHEADS 16
#define HDIM   64
#define BATCH  2
#define SEQ    2048
#define MTOT   (BATCH * SEQ)   // 4096 rows

// ---- CDNA5 async-to-LDS DMA (ASYNCcnt path), portable via inline asm --------
static __device__ __forceinline__ void async_ld_b128(uint32_t lds_off, const void* g) {
  asm volatile("global_load_async_to_lds_b128 %0, %1, off"
               :: "v"(lds_off), "v"(g) : "memory");
}
#define S_WAIT_ASYNC(nstr) asm volatile("s_wait_asynccnt " nstr ::: "memory")

// LDS byte offset of a __shared__ object = low 32 bits of its generic address.
static __device__ __forceinline__ uint32_t lds_off(const void* p) {
  return (uint32_t)(uintptr_t)p;
}

// Build a 16x32 f16 A-fragment for v_wmma_f32_16x16x32_f16.
// Layout: lanes 0-15 hold K=0..15, lanes 16-31 hold K=16..31 (matches B fills).
static __device__ __forceinline__ v16h load_afrag(const f16* base, int hi) {
  v8h lo  = *(const v8h*)(base + hi * 16);
  v8h hi8 = *(const v8h*)(base + hi * 16 + 8);
  return __builtin_shufflevector(lo, hi8, 0, 1, 2, 3, 4, 5, 6, 7,
                                          8, 9, 10, 11, 12, 13, 14, 15);
}

static __device__ __forceinline__ v8f wmma16(v16h a, v16h b, v8f c) {
  return __builtin_amdgcn_wmma_f32_16x16x32_f16(false, a, false, b,
                                                (short)0, c, false, false);
}

// ---------------------------------------------------------------- fp32 -> f16
__global__ void k_cvt(const float* __restrict__ src, f16* __restrict__ dst, int n) {
  int i = blockIdx.x * blockDim.x + threadIdx.x;
  int stride = gridDim.x * blockDim.x;
  for (; i < n; i += stride) dst[i] = (f16)src[i];
}

// --------------------------------------------- fused QKV projection + norm/rope
// grid = (32, 16, 3). Each wave computes a 32x64 tile; the 64x64 weight tile
// (shared by all 4 waves) is double-buffered in LDS via async DMA.
__global__ void __launch_bounds__(128) k_proj(
    const f16* __restrict__ X,
    const f16* __restrict__ Wq, const f16* __restrict__ Wk, const f16* __restrict__ Wv,
    const float* __restrict__ bq, const float* __restrict__ bk, const float* __restrict__ bv,
    const float* __restrict__ qnw, const float* __restrict__ knw,
    const float* __restrict__ ropec, const float* __restrict__ ropes,
    f16* __restrict__ Q, f16* __restrict__ K, f16* __restrict__ Vt) {
  __shared__ __align__(128) f16 wtile[2][64][64];   // [buf][out-col][k]  16 KB

  const int tid = threadIdx.x;
  const int lane = tid & 31, wave = tid >> 5;
  const int ln = lane & 15, hi = lane >> 4;
  const int m0 = blockIdx.x * 128 + wave * 32;
  const int h  = blockIdx.y;
  const int mode = blockIdx.z;        // 0=q 1=k 2=v

  const f16* W      = (mode == 0) ? Wq : (mode == 1) ? Wk : Wv;
  const float* bias = (mode == 0) ? bq : (mode == 1) ? bk : bv;

  // stage one 64x64 weight tile: 512 x 16B chunks, 4 per thread
  auto stage = [&](int buf, int k0) {
#pragma unroll
    for (int e = 0; e < 4; ++e) {
      const int c = tid * 4 + e;
      const int row = c >> 3, q = c & 7;
      async_ld_b128(lds_off(&wtile[buf][row][q * 8]),
                    W + (size_t)(h * 64 + row) * DIM + k0 + q * 8);
    }
  };

  stage(0, 0);

  v8f acc[2][4] = {};
  for (int k0 = 0; k0 < DIM; k0 += 64) {
    const int buf = (k0 >> 6) & 1;
    __syncthreads();                       // all waves done reading buf^1
    if (k0 + 64 < DIM) { stage(buf ^ 1, k0 + 64); S_WAIT_ASYNC("0x4"); }
    else               { S_WAIT_ASYNC("0x0"); }
    __syncthreads();                       // current buffer valid for all waves

    if (k0 + 128 < DIM) __builtin_prefetch(X + (size_t)(m0 + ln) * DIM + k0 + 128, 0, 1);
#pragma unroll
    for (int kk = 0; kk < 64; kk += 32) {
      v16h a0 = load_afrag(X + (size_t)(m0 + ln) * DIM + k0 + kk, hi);
      v16h a1 = load_afrag(X + (size_t)(m0 + 16 + ln) * DIM + k0 + kk, hi);
#pragma unroll
      for (int j = 0; j < 4; ++j) {
        v16h b = *(const v16h*)(&wtile[buf][j * 16 + ln][kk + hi * 16]);
        acc[0][j] = wmma16(a0, b, acc[0][j]);
        acc[1][j] = wmma16(a1, b, acc[1][j]);
      }
    }
  }

  const int b = m0 >> 11;                  // m0 / SEQ (blocks never straddle)
  const int srowbase = m0 & (SEQ - 1);

  float y[2][4][8];
#pragma unroll
  for (int g = 0; g < 2; ++g)
#pragma unroll
    for (int j = 0; j < 4; ++j) {
      float bb = bias[h * 64 + j * 16 + ln];
#pragma unroll
      for (int v = 0; v < 8; ++v) y[g][j][v] = acc[g][j][v] + bb;
    }

  if (mode < 2) {
    const float* nw = (mode == 0) ? qnw : knw;
    float wcol[4];
#pragma unroll
    for (int j = 0; j < 4; ++j) wcol[j] = nw[j * 16 + ln];
#pragma unroll
    for (int g = 0; g < 2; ++g)
#pragma unroll
      for (int v = 0; v < 8; ++v) {
        float ssq = 0.f;
#pragma unroll
        for (int j = 0; j < 4; ++j) ssq += y[g][j][v] * y[g][j][v];
        for (int o = 1; o < 16; o <<= 1) ssq += __shfl_xor(ssq, o, 32);
        float inv = rsqrtf(ssq * (1.0f / HDIM) + 1e-6f);
#pragma unroll
        for (int j = 0; j < 4; ++j) y[g][j][v] *= inv * wcol[j];
      }
    // RoPE: interleaved pair lives in adjacent lane
#pragma unroll
    for (int j = 0; j < 4; ++j) {
      const int col = j * 16 + ln;
      const int fr  = col >> 1;
      const float sgn = (col & 1) ? 1.0f : -1.0f;
#pragma unroll
      for (int g = 0; g < 2; ++g)
#pragma unroll
        for (int v = 0; v < 8; ++v) {
          const int srow = srowbase + g * 16 + v + hi * 8;
          const size_t ri = ((size_t)b * SEQ + srow) * (HDIM / 2) + fr;
          float c = ropec[ri], s = ropes[ri];
          float part = __shfl_xor(y[g][j][v], 1, 32);
          y[g][j][v] = y[g][j][v] * c + sgn * part * s;
        }
    }
    f16* out = (mode == 0) ? Q : K;
#pragma unroll
    for (int g = 0; g < 2; ++g)
#pragma unroll
      for (int j = 0; j < 4; ++j)
#pragma unroll
        for (int v = 0; v < 8; ++v) {
          const int srow = srowbase + g * 16 + v + hi * 8;
          out[(((size_t)b * NHEADS + h) * SEQ + srow) * HDIM + j * 16 + ln] =
              (f16)y[g][j][v];
        }
  } else {
#pragma unroll
    for (int g = 0; g < 2; ++g)
#pragma unroll
      for (int j = 0; j < 4; ++j)
#pragma unroll
        for (int v = 0; v < 8; ++v) {
          const int srow = srowbase + g * 16 + v + hi * 8;
          Vt[(((size_t)b * NHEADS + h) * HDIM + j * 16 + ln) * SEQ + srow] =
              (f16)y[g][j][v];
        }
  }
}

// ------------------------------------------------------- flash attention core
// One wave per 16-query tile; 4 waves share (b,h). 64 keys per iteration:
// 64x64 K tile + 64x64 V^T tile double-buffered in LDS via async DMA.
__global__ void __launch_bounds__(128) k_attn(
    const f16* __restrict__ Q, const f16* __restrict__ K,
    const f16* __restrict__ Vt, f16* __restrict__ O) {
  __shared__ __align__(128) f16 ktile[2][64][64];   // [buf][key][d]    16 KB
  __shared__ __align__(128) f16 vtile[2][64][64];   // [buf][d][key]    16 KB
  __shared__ __align__(64)  f16 pbuf[4][16][64];    // per-wave P        8 KB

  const int tid = threadIdx.x;
  const int lane = tid & 31, wave = tid >> 5;
  const int ln = lane & 15, hi = lane >> 4;
  const int id = blockIdx.x * 4 + wave;
  const int qt = id & 127;
  const int bh = id >> 7;
  const int qs = qt * 16;

  const f16* kbase = K + (size_t)bh * SEQ * HDIM;
  const f16* vbase = Vt + (size_t)bh * HDIM * SEQ;

  // stage 64x64 K tile + 64x64 V tile: 1024 x 16B chunks, 8 per thread
  auto stage = [&](int buf, int kt) {
#pragma unroll
    for (int e = 0; e < 4; ++e) {
      const int c = tid * 4 + e;
      const int row = c >> 3, q = c & 7;
      async_ld_b128(lds_off(&ktile[buf][row][q * 8]),
                    kbase + (size_t)(kt + row) * HDIM + q * 8);
      async_ld_b128(lds_off(&vtile[buf][row][q * 8]),
                    vbase + (size_t)row * SEQ + kt + q * 8);
    }
  };

  const f16* qrow = Q + ((size_t)bh * SEQ + qs + ln) * HDIM;
  const v16h a0 = load_afrag(qrow, hi);
  const v16h a1 = load_afrag(qrow + 32, hi);

  stage(0, 0);

  v8f oa[4] = {};
  float m[8], l[8];
#pragma unroll
  for (int v = 0; v < 8; ++v) { m[v] = -1e30f; l[v] = 0.f; }
  const float sc = 0.125f;

  for (int kt = 0; kt < SEQ; kt += 64) {
    const int buf = (kt >> 6) & 1;
    __syncthreads();
    if (kt + 64 < SEQ) { stage(buf ^ 1, kt + 64); S_WAIT_ASYNC("0x8"); }
    else               { S_WAIT_ASYNC("0x0"); }
    __syncthreads();

    // scores: 4 tiles of 16 keys, contraction over d=64 in two chunks
    v8f s[4] = {};
#pragma unroll
    for (int t = 0; t < 4; ++t) {
      v16h b0 = *(const v16h*)(&ktile[buf][t * 16 + ln][hi * 16]);
      v16h b1 = *(const v16h*)(&ktile[buf][t * 16 + ln][32 + hi * 16]);
      s[t] = wmma16(a0, b0, s[t]);
      s[t] = wmma16(a1, b1, s[t]);
    }

    float p[4][8], mn[8];
#pragma unroll
    for (int v = 0; v < 8; ++v) {
      float rm = -1e30f;
#pragma unroll
      for (int t = 0; t < 4; ++t) {
        float x = s[t][v] * sc;
        s[t][v] = x;
        rm = fmaxf(rm, x);
      }
      for (int o = 1; o < 16; o <<= 1) rm = fmaxf(rm, __shfl_xor(rm, o, 32));
      mn[v] = fmaxf(m[v], rm);
    }
#pragma unroll
    for (int v = 0; v < 8; ++v) {
      float alpha = __expf(m[v] - mn[v]);
      float rs = 0.f;
#pragma unroll
      for (int t = 0; t < 4; ++t) {
        p[t][v] = __expf(s[t][v] - mn[v]);
        rs += p[t][v];
      }
      for (int o = 1; o < 16; o <<= 1) rs += __shfl_xor(rs, o, 32);
      l[v] = l[v] * alpha + rs;
      m[v] = mn[v];
#pragma unroll
      for (int j = 0; j < 4; ++j) oa[j][v] *= alpha;
    }

    // C-layout -> A-layout transpose of P (same-wave region, DS in-order)
#pragma unroll
    for (int v = 0; v < 8; ++v) {
      const int row = v + hi * 8;
#pragma unroll
      for (int t = 0; t < 4; ++t)
        pbuf[wave][row][t * 16 + ln] = (f16)p[t][v];
    }
    asm volatile("s_wait_dscnt 0" ::: "memory");
    v16h pf0 = load_afrag(&pbuf[wave][ln][0], hi);    // keys 0..31
    v16h pf1 = load_afrag(&pbuf[wave][ln][32], hi);   // keys 32..63

#pragma unroll
    for (int j = 0; j < 4; ++j) {
      v16h vb0 = *(const v16h*)(&vtile[buf][j * 16 + ln][hi * 16]);
      v16h vb1 = *(const v16h*)(&vtile[buf][j * 16 + ln][32 + hi * 16]);
      oa[j] = wmma16(pf0, vb0, oa[j]);
      oa[j] = wmma16(pf1, vb1, oa[j]);
    }
  }

  const int b = bh >> 4, h = bh & 15;
#pragma unroll
  for (int v = 0; v < 8; ++v) {
    const float invl = 1.0f / l[v];
    const int srow = qs + v + hi * 8;
#pragma unroll
    for (int j = 0; j < 4; ++j)
      O[((size_t)b * SEQ + srow) * DIM + h * 64 + j * 16 + ln] =
          (f16)(oa[j][v] * invl);
  }
}

// ------------------------------------------------------------ output projection
__global__ void __launch_bounds__(128) k_oproj(
    const f16* __restrict__ X, const f16* __restrict__ W,
    const float* __restrict__ bias, float* __restrict__ out) {
  __shared__ __align__(128) f16 wtile[2][64][64];

  const int tid = threadIdx.x;
  const int lane = tid & 31, wave = tid >> 5;
  const int ln = lane & 15, hi = lane >> 4;
  const int m0 = blockIdx.x * 128 + wave * 32;
  const int n0 = blockIdx.y * 64;

  auto stage = [&](int buf, int k0) {
#pragma unroll
    for (int e = 0; e < 4; ++e) {
      const int c = tid * 4 + e;
      const int row = c >> 3, q = c & 7;
      async_ld_b128(lds_off(&wtile[buf][row][q * 8]),
                    W + (size_t)(n0 + row) * DIM + k0 + q * 8);
    }
  };

  stage(0, 0);

  v8f acc[2][4] = {};
  for (int k0 = 0; k0 < DIM; k0 += 64) {
    const int buf = (k0 >> 6) & 1;
    __syncthreads();
    if (k0 + 64 < DIM) { stage(buf ^ 1, k0 + 64); S_WAIT_ASYNC("0x4"); }
    else               { S_WAIT_ASYNC("0x0"); }
    __syncthreads();

    if (k0 + 128 < DIM) __builtin_prefetch(X + (size_t)(m0 + ln) * DIM + k0 + 128, 0, 1);
#pragma unroll
    for (int kk = 0; kk < 64; kk += 32) {
      v16h a0 = load_afrag(X + (size_t)(m0 + ln) * DIM + k0 + kk, hi);
      v16h a1 = load_afrag(X + (size_t)(m0 + 16 + ln) * DIM + k0 + kk, hi);
#pragma unroll
      for (int j = 0; j < 4; ++j) {
        v16h b = *(const v16h*)(&wtile[buf][j * 16 + ln][kk + hi * 16]);
        acc[0][j] = wmma16(a0, b, acc[0][j]);
        acc[1][j] = wmma16(a1, b, acc[1][j]);
      }
    }
  }
#pragma unroll
  for (int g = 0; g < 2; ++g)
#pragma unroll
    for (int j = 0; j < 4; ++j) {
      const float bb = bias[n0 + j * 16 + ln];
#pragma unroll
      for (int v = 0; v < 8; ++v)
        out[(size_t)(m0 + g * 16 + v + hi * 8) * DIM + n0 + j * 16 + ln] =
            acc[g][j][v] + bb;
    }
}

// ---------------------------------------------------------------------- launch
extern "C" void kernel_launch(void* const* d_in, const int* in_sizes, int n_in,
                              void* d_out, int out_size, void* d_ws, size_t ws_size,
                              hipStream_t stream) {
  (void)in_sizes; (void)n_in; (void)out_size; (void)ws_size;
  const float* hs    = (const float*)d_in[0];
  const float* ropec = (const float*)d_in[1];
  const float* ropes = (const float*)d_in[2];
  const float* Wq = (const float*)d_in[3];  const float* bq = (const float*)d_in[4];
  const float* Wk = (const float*)d_in[5];  const float* bk = (const float*)d_in[6];
  const float* Wv = (const float*)d_in[7];  const float* bv = (const float*)d_in[8];
  const float* qnw = (const float*)d_in[9]; const float* knw = (const float*)d_in[10];
  const float* Wo = (const float*)d_in[11]; const float* bo = (const float*)d_in[12];
  float* out = (float*)d_out;

  char* ws = (char*)d_ws;
  f16* X16  = (f16*)(ws);                       //  8 MB  [M, D]
  f16* Wq16 = (f16*)(ws + (size_t)( 8 << 20));  //  2 MB
  f16* Wk16 = (f16*)(ws + (size_t)(10 << 20));  //  2 MB
  f16* Wv16 = (f16*)(ws + (size_t)(12 << 20));  //  2 MB
  f16* Wo16 = (f16*)(ws + (size_t)(14 << 20));  //  2 MB
  f16* Q16  = (f16*)(ws + (size_t)(16 << 20));  //  8 MB  [B,H,S,hd]
  f16* K16  = (f16*)(ws + (size_t)(24 << 20));  //  8 MB  [B,H,S,hd]
  f16* Vt16 = (f16*)(ws + (size_t)(32 << 20));  //  8 MB  [B,H,hd,S]
  f16* O16  = (f16*)(ws + (size_t)(40 << 20));  //  8 MB  [B,S,D]

  k_cvt<<<1024, 256, 0, stream>>>(hs, X16, MTOT * DIM);
  k_cvt<<<512, 256, 0, stream>>>(Wq, Wq16, DIM * DIM);
  k_cvt<<<512, 256, 0, stream>>>(Wk, Wk16, DIM * DIM);
  k_cvt<<<512, 256, 0, stream>>>(Wv, Wv16, DIM * DIM);
  k_cvt<<<512, 256, 0, stream>>>(Wo, Wo16, DIM * DIM);

  k_proj<<<dim3(MTOT / 128, NHEADS, 3), 128, 0, stream>>>(
      X16, Wq16, Wk16, Wv16, bq, bk, bv, qnw, knw, ropec, ropes, Q16, K16, Vt16);

  k_attn<<<dim3((BATCH * NHEADS * (SEQ / 16)) / 4), 128, 0, stream>>>(
      Q16, K16, Vt16, O16);

  k_oproj<<<dim3(MTOT / 128, DIM / 64), 128, 0, stream>>>(O16, Wo16, bo, out);
}
